// EdgeGenerator_58308476011007
// MI455X (gfx1250) — matrix operand order repaired and checked by reading, not compiled, not run on hardware
//
#include <hip/hip_runtime.h>

#define N_NODES 10000
#define N_EDGES 100000
#define D_IN    2560
#define H1      512
#define H2      256

typedef __attribute__((ext_vector_type(16))) __bf16        v16bf;
typedef __attribute__((ext_vector_type(8)))  float         v8f;
typedef __attribute__((ext_vector_type(8)))  unsigned int  v8u;

__device__ __forceinline__ unsigned short bf_bits(float f) {
    unsigned u = __builtin_bit_cast(unsigned, f);
    unsigned r = u + 0x7FFFu + ((u >> 16) & 1u);   // round-to-nearest-even
    return (unsigned short)(r >> 16);
}

// CDNA5 async global->LDS copy (16B per lane), tracked by ASYNCcnt.
// LDS operand is the wave-relative LDS byte address = low 32 bits of the
// generic pointer (ISA 10.2).
__device__ __forceinline__ void async_copy16(void* lds_dst, const void* gsrc) {
    unsigned ldsoff = (unsigned)(size_t)lds_dst;
    asm volatile("global_load_async_to_lds_b128 %0, %1, off"
                 :: "v"(ldsoff), "v"(gsrc) : "memory");
}
__device__ __forceinline__ void wait_async0() {
    asm volatile("s_wait_asynccnt 0x0" ::: "memory");
}

// ---------------- prep kernels ----------------

__global__ void zero_init_kernel(int* deg, float* kld_out) {
    int i = blockIdx.x * blockDim.x + threadIdx.x;
    if (i < N_NODES) deg[i] = 0;
    if (i == 0) *kld_out = 0.0f;
}

__global__ void count_deg_kernel(const int* __restrict__ ecol, int* __restrict__ deg) {
    int e = blockIdx.x * blockDim.x + threadIdx.x;
    if (e < N_EDGES) atomicAdd(&deg[ecol[e]], 1);
}

__global__ void dinv_kernel(const int* __restrict__ deg, float* __restrict__ dinv) {
    int i = blockIdx.x * blockDim.x + threadIdx.x;
    if (i < N_NODES) dinv[i] = rsqrtf((float)deg[i] + 1.0f);   // +1 = self loop
}

__global__ void cvt_x_kernel(const float* __restrict__ x, unsigned int* __restrict__ xbf2,
                             int n2) {
    int i = blockIdx.x * blockDim.x + threadIdx.x;
    if (i >= n2) return;
    float2 f = ((const float2*)x)[i];
    xbf2[i] = (unsigned)bf_bits(f.x) | ((unsigned)bf_bits(f.y) << 16);
}

// Repack W (row-major KxN f32) into per-fragment contiguous bf16:
// offset = ((ntile*ksteps + kstep)*32 + lane)*16 + j, with
// n = ntile*16 + (lane&15), k = kstep*32 + (lane>>4)*16 + j   (contiguous k order, same as the A fragments)
__global__ void pack_w_kernel(const float* __restrict__ W, unsigned short* __restrict__ Wp,
                              int K, int N) {
    int o = blockIdx.x * blockDim.x + threadIdx.x;
    if (o >= K * N) return;
    int ksteps = K >> 5;
    int j     = o & 15;
    int lane  = (o >> 4) & 31;
    int tile  = o >> 9;
    int kstep = tile % ksteps;
    int ntile = tile / ksteps;
    int n = ntile * 16 + (lane & 15);
    int k = kstep * 32 + (lane >> 4) * 16 + j;
    Wp[o] = bf_bits(W[(size_t)k * N + n]);
}

// ---------------- GEMM layer 1 (fused edge-gather) ----------------
// A[e, 0:1280] = x[rows[e]], A[e, 1280:2560] = x[cols[e]]   (bf16)
// g1 = A @ W1 -> [E, 512] f32
// Block: 32-row M-tile x full N=512; 8 waves, each wave: 2 M-subtiles x 4 N-tiles.
// A staged in double-buffered LDS via async global->LDS copies.
__global__ __launch_bounds__(256) void gemm1_kernel(
    const unsigned short* __restrict__ xbf,
    const int* __restrict__ erow, const int* __restrict__ ecol,
    const unsigned short* __restrict__ W1p,
    float* __restrict__ g1)
{
    __shared__ unsigned int As32[2][32 * 64];   // 2 x (32 rows x 128 bf16)
    __shared__ int s_node[64];                  // [0:32)=rows, [32:64)=cols

    const int tid = threadIdx.x;
    const int e0  = blockIdx.x * 32;
    if (tid < 32)       s_node[tid] = erow[e0 + tid];
    else if (tid < 64)  s_node[tid] = ecol[e0 + (tid - 32)];
    __syncthreads();

    const int wave = tid >> 5, lane = tid & 31;
    const int half = lane >> 4, l16 = lane & 15;

    v8f acc[2][4] = {{v8f{}, v8f{}, v8f{}, v8f{}}, {v8f{}, v8f{}, v8f{}, v8f{}}};

    // stage chunk c (20 chunks of 128 bf16: 10 from row node, 10 from col node)
    auto stage = [&](int buf, int c) {
        #pragma unroll
        for (int q = 0; q < 2; ++q) {
            int idx = tid + q * 256;            // 0..511 -> 32 rows x 16 chunks of 16B
            int row = idx >> 4, sub = idx & 15;
            int node = (c < 10) ? s_node[row] : s_node[32 + row];
            const unsigned short* src =
                xbf + (size_t)node * 1280 + (c < 10 ? c : c - 10) * 128 + sub * 8;
            async_copy16(&As32[buf][row * 64 + sub * 4], src);
        }
    };

    stage(0, 0);
    for (int c = 0; c < 20; ++c) {
        const int buf = c & 1;
        wait_async0();
        __syncthreads();                         // buffer `buf` visible to all waves
        if (c + 1 < 20) stage(buf ^ 1, c + 1);   // overlap next fill with compute

        #pragma unroll
        for (int kk = 0; kk < 4; ++kk) {         // 4 wmma K-steps of 32
            v16bf a[2];
            #pragma unroll
            for (int mt = 0; mt < 2; ++mt) {     // 16-bit A 16x32, contiguous k = 16*half + s
                v8u au;
                #pragma unroll
                for (int v = 0; v < 8; ++v) {
                    int kp2 = kk * 16 + half * 8 + v;
                    au[v] = As32[buf][(mt * 16 + l16) * 64 + kp2];
                }
                a[mt] = __builtin_bit_cast(v16bf, au);
            }
            const int kstep = c * 4 + kk;        // 0..79
            #pragma unroll
            for (int t = 0; t < 4; ++t) {
                const int ntile = wave * 4 + t;  // 0..31
                v8u bu = *(const v8u*)((const unsigned int*)W1p +
                                       (size_t)((ntile * 80 + kstep) * 32 + lane) * 8);
                v16bf b = __builtin_bit_cast(v16bf, bu);
                acc[0][t] = __builtin_amdgcn_wmma_f32_16x16x32_bf16(
                                false, a[0], false, b, (short)0, acc[0][t], false, false);
                acc[1][t] = __builtin_amdgcn_wmma_f32_16x16x32_bf16(
                                false, a[1], false, b, (short)0, acc[1][t], false, false);
            }
        }
    }

    #pragma unroll
    for (int mt = 0; mt < 2; ++mt)
        #pragma unroll
        for (int t = 0; t < 4; ++t) {
            const int n = (wave * 4 + t) * 16 + l16;
            #pragma unroll
            for (int v = 0; v < 8; ++v)          // C/D: M = half*8+v, N = lane&15
                g1[(size_t)(e0 + mt * 16 + half * 8 + v) * H1 + n] = acc[mt][t][v];
        }
}

// ---------------- GEMM layer 2 (dense) ----------------
// g2 = h1bf[E,512] @ W2 -> [E,256] f32
__global__ __launch_bounds__(256) void gemm2_kernel(
    const unsigned short* __restrict__ h1bf,
    const unsigned short* __restrict__ W2p,
    float* __restrict__ g2)
{
    __shared__ unsigned int As32[2][32 * 64];
    const int tid = threadIdx.x;
    const int e0  = blockIdx.x * 32;
    const int wave = tid >> 5, lane = tid & 31;
    const int half = lane >> 4, l16 = lane & 15;

    v8f acc[2][2] = {{v8f{}, v8f{}}, {v8f{}, v8f{}}};

    auto stage = [&](int buf, int c) {           // K = 512 -> 4 chunks of 128
        #pragma unroll
        for (int q = 0; q < 2; ++q) {
            int idx = tid + q * 256;
            int row = idx >> 4, sub = idx & 15;
            const unsigned short* src =
                h1bf + (size_t)(e0 + row) * H1 + c * 128 + sub * 8;
            async_copy16(&As32[buf][row * 64 + sub * 4], src);
        }
    };

    stage(0, 0);
    for (int c = 0; c < 4; ++c) {
        const int buf = c & 1;
        wait_async0();
        __syncthreads();
        if (c + 1 < 4) stage(buf ^ 1, c + 1);

        #pragma unroll
        for (int kk = 0; kk < 4; ++kk) {
            v16bf a[2];
            #pragma unroll
            for (int mt = 0; mt < 2; ++mt) {
                v8u au;
                #pragma unroll
                for (int v = 0; v < 8; ++v) {
                    int kp2 = kk * 16 + half * 8 + v;
                    au[v] = As32[buf][(mt * 16 + l16) * 64 + kp2];
                }
                a[mt] = __builtin_bit_cast(v16bf, au);
            }
            const int kstep = c * 4 + kk;        // 0..15
            #pragma unroll
            for (int t = 0; t < 2; ++t) {
                const int ntile = wave * 2 + t;  // 0..15
                v8u bu = *(const v8u*)((const unsigned int*)W2p +
                                       (size_t)((ntile * 16 + kstep) * 32 + lane) * 8);
                v16bf b = __builtin_bit_cast(v16bf, bu);
                acc[0][t] = __builtin_amdgcn_wmma_f32_16x16x32_bf16(
                                false, a[0], false, b, (short)0, acc[0][t], false, false);
                acc[1][t] = __builtin_amdgcn_wmma_f32_16x16x32_bf16(
                                false, a[1], false, b, (short)0, acc[1][t], false, false);
            }
        }
    }

    #pragma unroll
    for (int mt = 0; mt < 2; ++mt)
        #pragma unroll
        for (int t = 0; t < 2; ++t) {
            const int n = (wave * 2 + t) * 16 + l16;
            #pragma unroll
            for (int v = 0; v < 8; ++v)
                g2[(size_t)(e0 + mt * 16 + half * 8 + v) * H2 + n] = acc[mt][t][v];
        }
}

// ---------------- GCN scatter ----------------

// out[i] = g[i] * (i < N_NODES ? dinv[i]^2 : 1)   (self-loop term; also zero-inits accum)
__global__ void init_out_kernel(const float* __restrict__ g, const float* __restrict__ dinv,
                                float* __restrict__ out, size_t total, int shift) {
    size_t i = (size_t)blockIdx.x * blockDim.x + threadIdx.x;
    if (i >= total) return;
    size_t row = i >> shift;
    float d = (row < N_NODES) ? dinv[row] : 1.0f;
    out[i] = g[i] * d * d;
}

// out[col[e]][:] += g[row[e]][:] * dinv[row]*dinv[col]   (edge messages)
__global__ __launch_bounds__(256) void scatter_edges_kernel(
    const float* __restrict__ g, const int* __restrict__ erow, const int* __restrict__ ecol,
    const float* __restrict__ dinv, float* __restrict__ out, int ncols) {
    const int e = blockIdx.x;
    const int r = erow[e], c = ecol[e];
    const float w = dinv[r] * dinv[c];
    const float* src = g + (size_t)r * ncols;
    float* dst = out + (size_t)c * ncols;
    for (int j = threadIdx.x; j < ncols; j += blockDim.x)
        atomicAdd(dst + j, src[j] * w);
}

// h = relu(acc + bias) -> bf16 (feeds next WMMA layer)
__global__ void finalize_bf_kernel(const float* __restrict__ acc, const float* __restrict__ bias,
                                   unsigned short* __restrict__ hbf, size_t total, int mask) {
    size_t i = (size_t)blockIdx.x * blockDim.x + threadIdx.x;
    if (i >= total) return;
    float v = acc[i] + bias[(int)(i & mask)];
    hbf[i] = bf_bits(fmaxf(v, 0.0f));
}

// h = relu(acc + bias) -> f32 (final layer input)
__global__ void finalize_f32_kernel(const float* __restrict__ acc, const float* __restrict__ bias,
                                    float* __restrict__ h, size_t total, int mask) {
    size_t i = (size_t)blockIdx.x * blockDim.x + threadIdx.x;
    if (i >= total) return;
    h[i] = fmaxf(acc[i] + bias[(int)(i & mask)], 0.0f);
}

// ---------------- head: pre = h2 @ Wl + bl; gumbel-sigmoid; KLD mean ----------------
__global__ __launch_bounds__(256) void final_kernel(
    const float* __restrict__ h2, const float* __restrict__ Wl, const float* __restrict__ bl,
    const float* __restrict__ eps, float* __restrict__ d_out) {
    __shared__ float s_wl[H2];
    __shared__ float s_kld[8];
    const int tid = threadIdx.x;
    s_wl[tid] = Wl[tid];
    __syncthreads();

    const int wave = tid >> 5, lane = tid & 31;
    const int e = blockIdx.x * 8 + wave;       // one wave32 per edge
    const float* row = h2 + (size_t)e * H2;
    float p = 0.0f;
    #pragma unroll
    for (int j = 0; j < 8; ++j) p += row[lane + j * 32] * s_wl[lane + j * 32];
    #pragma unroll
    for (int off = 16; off > 0; off >>= 1) p += __shfl_down(p, off, 32);

    if (lane == 0) {
        float pre  = p + bl[0];
        float ep   = eps[e];
        float gate = logf(ep) - log1pf(-ep) + pre;
        float m    = 1.0f / (1.0f + expf(-gate));
        d_out[1 + e] = m;
        float pos = m, neg = 1.0f - m;
        s_kld[wave] = pos * logf(pos * 2.0f + 1e-8f) + neg * logf(neg * 2.0f + 1e-9f);
    }
    __syncthreads();
    if (tid == 0) {
        float s = 0.0f;
        #pragma unroll
        for (int i = 0; i < 8; ++i) s += s_kld[i];
        atomicAdd(d_out, s * (1.0f / N_EDGES));
    }
}

// ---------------- launch ----------------

extern "C" void kernel_launch(void* const* d_in, const int* in_sizes, int n_in,
                              void* d_out, int out_size, void* d_ws, size_t ws_size,
                              hipStream_t stream) {
    const float* x    = (const float*)d_in[0];
    const int*   eidx = (const int*)d_in[1];
    const float* eps  = (const float*)d_in[2];
    const float* W1   = (const float*)d_in[3];
    const float* b1   = (const float*)d_in[4];
    const float* W2   = (const float*)d_in[5];
    const float* b2   = (const float*)d_in[6];
    const float* Wl   = (const float*)d_in[7];
    const float* bl   = (const float*)d_in[8];
    const int* erow = eidx;
    const int* ecol = eidx + N_EDGES;
    float* out = (float*)d_out;

    char* p = (char*)d_ws;
    size_t off = 0;
    auto alloc = [&](size_t bytes) -> void* {
        void* r = p + off;
        off = (off + bytes + 255) & ~(size_t)255;
        return r;
    };
    int*            deg  = (int*)           alloc((size_t)N_NODES * 4);
    float*          dinv = (float*)         alloc((size_t)N_NODES * 4);
    unsigned short* xbf  = (unsigned short*)alloc((size_t)N_NODES * 1280 * 2);
    unsigned short* W1p  = (unsigned short*)alloc((size_t)D_IN * H1 * 2);
    unsigned short* W2p  = (unsigned short*)alloc((size_t)H1 * H2 * 2);
    float*          bufA = (float*)         alloc((size_t)N_EDGES * H1 * 4);  // g1 -> g2
    float*          bufB = (float*)         alloc((size_t)N_EDGES * H1 * 4);  // out1 -> out2
    void*           bufC =                  alloc((size_t)N_EDGES * H1 * 2);  // h1bf -> h2f
    (void)ws_size; (void)n_in; (void)in_sizes; (void)out_size;

    float*          g1   = bufA;
    float*          out1 = bufB;
    unsigned short* h1bf = (unsigned short*)bufC;
    float*          g2   = bufA;
    float*          out2 = bufB;
    float*          h2f  = (float*)bufC;

    const size_t tot1 = (size_t)N_EDGES * H1;   // 51.2M
    const size_t tot2 = (size_t)N_EDGES * H2;   // 25.6M

    zero_init_kernel<<<(N_NODES + 255) / 256, 256, 0, stream>>>(deg, out);
    count_deg_kernel<<<(N_EDGES + 255) / 256, 256, 0, stream>>>(ecol, deg);
    dinv_kernel<<<(N_NODES + 255) / 256, 256, 0, stream>>>(deg, dinv);

    const int n2 = N_NODES * 1280 / 2;
    cvt_x_kernel<<<(n2 + 255) / 256, 256, 0, stream>>>(x, (unsigned int*)xbf, n2);
    pack_w_kernel<<<(D_IN * H1 + 255) / 256, 256, 0, stream>>>(W1, W1p, D_IN, H1);
    pack_w_kernel<<<(H1 * H2 + 255) / 256, 256, 0, stream>>>(W2, W2p, H1, H2);

    // layer 1
    gemm1_kernel<<<N_EDGES / 32, 256, 0, stream>>>(xbf, erow, ecol, W1p, g1);
    init_out_kernel<<<(unsigned)((tot1 + 255) / 256), 256, 0, stream>>>(g1, dinv, out1, tot1, 9);
    scatter_edges_kernel<<<N_EDGES, 256, 0, stream>>>(g1, erow, ecol, dinv, out1, H1);
    finalize_bf_kernel<<<(unsigned)((tot1 + 255) / 256), 256, 0, stream>>>(out1, b1, h1bf, tot1, H1 - 1);

    // layer 2
    gemm2_kernel<<<N_EDGES / 32, 256, 0, stream>>>(h1bf, W2p, g2);
    init_out_kernel<<<(unsigned)((tot2 + 255) / 256), 256, 0, stream>>>(g2, dinv, out2, tot2, 8);
    scatter_edges_kernel<<<N_EDGES, 256, 0, stream>>>(g2, erow, ecol, dinv, out2, H2);
    finalize_f32_kernel<<<(unsigned)((tot2 + 255) / 256), 256, 0, stream>>>(out2, b2, h2f, tot2, H2 - 1);

    // head
    final_kernel<<<N_EDGES / 8, 256, 0, stream>>>(h2f, Wl, bl, eps, out);
}
